// GSAPool_88656714924906
// MI455X (gfx1250) — matrix operand order repaired and checked by reading, not compiled, not run on hardware
//
#include <hip/hip_runtime.h>
#include <hip/hip_bf16.h>

// ---------------------------------------------------------------------------
// GSAPool for MI455X (gfx1250, wave32).
//   N=100000 nodes, E=1600000 edges, D=128, K=50000.
// x (51.2MB) and agg (25.6MB) are L2-resident (192MB L2) -> gather/scatter
// stages run at L2 bandwidth. Edge dot products batched 16-per-wave through
// V_WMMA_F32_16X16X32_BF16 (diagonal of the 16x16 product = 16 edge dots).
// ---------------------------------------------------------------------------

#define Nn 100000
#define Ee 1600000
#define Dd 128
#define Kk 50000
#define NBIN 65536

typedef __attribute__((ext_vector_type(16))) __bf16 v16bf;
typedef __attribute__((ext_vector_type(8)))  float  v8f;

// --- zero scratch accumulators --------------------------------------------
__global__ void k_zero(float* wsum, float* deg, unsigned* H) {
    int i = blockIdx.x * blockDim.x + threadIdx.x;
    if (i < Nn) { wsum[i] = 0.0f; deg[i] = 0.0f; }
    if (i < NBIN) H[i] = 0u;
}

// --- per-edge dot products via WMMA (16 edges per wave) --------------------
// A = rows x[row[eb..eb+15]] (16x32 bf16 chunk), B = rows x[col[...]] used as
// columns (32x16). Diagonal of A*B over 4 chunks = the 16 edge dots.
// A 16-bit layout: lanes0-15 row M=sub, v0-3 K=hi*8+0..7, v4-7 K=hi*8+16..23.
// B 16-bit layout: lane column N=sub, K = hi*8+0..7, hi*8+16..23 (same order as A).
__global__ void __launch_bounds__(256) k_edge_dot(
    const float* __restrict__ x, const int* __restrict__ row,
    const int* __restrict__ col, float* __restrict__ wsum,
    float* __restrict__ deg)
{
    const int lane = threadIdx.x & 31;
    const int wave = blockIdx.x * (blockDim.x >> 5) + (threadIdx.x >> 5);
    const int eb   = wave * 16;
    if (eb >= Ee) return;                 // wave-uniform: EXEC stays all-1s
    const int sub = lane & 15;
    const int hi  = lane >> 4;
    const int e   = eb + sub;
    const float* __restrict__ ar = x + (size_t)row[e] * Dd;
    const float* __restrict__ br = x + (size_t)col[e] * Dd;

    v8f c = {};
#pragma unroll
    for (int ch = 0; ch < 4; ++ch) {
        const int ka = ch * 32 + hi * 8;       // A: K = ka..ka+7, ka+16..ka+23
        float4 a0 = *(const float4*)(ar + ka + 0);
        float4 a1 = *(const float4*)(ar + ka + 4);
        float4 a2 = *(const float4*)(ar + ka + 16);
        float4 a3 = *(const float4*)(ar + ka + 20);
        const int kb = ch * 32 + hi * 8;       // B: K = kb..kb+7, kb+16..kb+23
        float4 b0 = *(const float4*)(br + kb + 0);
        float4 b1 = *(const float4*)(br + kb + 4);
        float4 b2 = *(const float4*)(br + kb + 16);
        float4 b3 = *(const float4*)(br + kb + 20);
        float af[16] = { a0.x,a0.y,a0.z,a0.w, a1.x,a1.y,a1.z,a1.w,
                         a2.x,a2.y,a2.z,a2.w, a3.x,a3.y,a3.z,a3.w };
        float bf[16] = { b0.x,b0.y,b0.z,b0.w, b1.x,b1.y,b1.z,b1.w,
                         b2.x,b2.y,b2.z,b2.w, b3.x,b3.y,b3.z,b3.w };
        v16bf av, bv;
#pragma unroll
        for (int i = 0; i < 16; ++i) { av[i] = (__bf16)af[i]; bv[i] = (__bf16)bf[i]; }
        c = __builtin_amdgcn_wmma_f32_16x16x32_bf16(
                false, av, false, bv, (short)0, c, false, false);
    }

    // diagonal: m<8 -> lane m, VGPR m ; m>=8 -> lane m+16, VGPR m-8
    if (lane < 8 || lane >= 24) {
        const int de  = (lane < 8) ? (eb + lane) : (eb + lane - 16);
        const int idx = (lane < 8) ? lane : (lane - 24);
        float w = c[0];
#pragma unroll
        for (int r = 1; r < 8; ++r) w = (idx == r) ? c[r] : w;
        const int rr = row[de];
        atomicAdd(&wsum[rr], w);
        atomicAdd(&deg[rr], 1.0f);
    }
}

// --- score + key + hi16 histogram -----------------------------------------
__global__ void k_score(const float* __restrict__ wsum,
                        const float* __restrict__ deg,
                        unsigned* __restrict__ key, unsigned* __restrict__ H)
{
    int i = blockIdx.x * blockDim.x + threadIdx.x;
    if (i >= Nn) return;
    float dg  = deg[i];
    float inv = dg > 0.0f ? 1.0f / dg : 0.0f;
    float s   = __expf(-wsum[i] * inv);          // always > 0
    unsigned k = __float_as_uint(s);             // positive floats: bits order == value order
    key[i] = k;
    atomicAdd(&H[k >> 16], 1u);
}

// --- exclusive scan of the 65536-bin histogram (single workgroup) ----------
__global__ void __launch_bounds__(1024) k_scan(const unsigned* __restrict__ H,
                                               unsigned* __restrict__ Epre,
                                               unsigned* __restrict__ cursor)
{
    __shared__ unsigned part[1024];
    const int t = threadIdx.x;
    unsigned s = 0;
    for (int i = 0; i < 64; ++i) s += H[t * 64 + i];
    part[t] = s;
    __syncthreads();
    for (int off = 1; off < 1024; off <<= 1) {
        unsigned v = (t >= off) ? part[t - off] : 0u;
        __syncthreads();
        part[t] += v;
        __syncthreads();
    }
    unsigned run = part[t] - s;                  // exclusive prefix of this chunk
    for (int i = 0; i < 64; ++i) {
        int b = t * 64 + i;
        Epre[b] = run; cursor[b] = run;
        run += H[b];
    }
}

// --- scatter node ids into their hi16 bin ---------------------------------
__global__ void k_scatter(const unsigned* __restrict__ key,
                          unsigned* __restrict__ cursor, int* __restrict__ binList)
{
    int i = blockIdx.x * blockDim.x + threadIdx.x;
    if (i >= Nn) return;
    unsigned b = key[i] >> 16;
    binList[atomicAdd(&cursor[b], 1u)] = i;
}

// --- exact rank (score desc, index asc) -> pos / perm ----------------------
__global__ void k_rank(const unsigned* __restrict__ key,
                       const unsigned* __restrict__ H,
                       const unsigned* __restrict__ Epre,
                       const int* __restrict__ binList,
                       const int* __restrict__ batch,
                       int* __restrict__ pos, int* __restrict__ perm_i,
                       float* __restrict__ o_perm, float* __restrict__ o_batch)
{
    int i = blockIdx.x * blockDim.x + threadIdx.x;
    if (i >= Nn) return;
    unsigned ki = key[i];
    unsigned b  = ki >> 16;
    unsigned start = Epre[b], cnt = H[b];
    unsigned r = (unsigned)Nn - start - cnt;     // keys in strictly-higher bins
    for (unsigned t = 0; t < cnt; ++t) {
        int j = binList[start + t];
        unsigned kj = key[j];
        if (kj > ki || (kj == ki && j < i)) ++r;
    }
    if (r < (unsigned)Kk) {
        pos[i]     = (int)r;
        perm_i[r]  = i;
        o_perm[r]  = (float)i;
        o_batch[r] = (float)batch[i];
    } else {
        pos[i] = -1;
    }
}

// --- seed agg with the self-loop contribution: agg[r] = x[perm[r]] --------
__global__ void k_agg_init(const float* __restrict__ x,
                           const int* __restrict__ perm_i, float* __restrict__ agg)
{
    int t = blockIdx.x * blockDim.x + threadIdx.x;
    if (t >= Kk * Dd) return;
    int r = t >> 7, d = t & 127;
    agg[t] = x[(size_t)perm_i[r] * Dd + d];
}

// --- edge aggregation: agg[pos[row]] += x[col], staged via async LDS copy --
__global__ void __launch_bounds__(256) k_agg_edges(
    const float* __restrict__ x, const int* __restrict__ row,
    const int* __restrict__ col, const int* __restrict__ pos,
    float* __restrict__ agg)
{
    __shared__ __align__(16) float stage[8 * Dd];
    const int lane = threadIdx.x & 31;
    const int w    = threadIdx.x >> 5;
    const int e    = blockIdx.x * 8 + w;
    if (e >= Ee) return;                    // wave-uniform
    const int p = pos[row[e]];
    if (p < 0) return;                      // wave-uniform
    const float* src = x + (size_t)col[e] * Dd;
    float* slot = &stage[w * Dd];
    // CDNA5 async copy: each lane moves 16B global -> LDS (ASYNCcnt tracked).
    unsigned lds_off  = (unsigned)(size_t)(slot + lane * 4);  // low 32 bits == LDS offset
    const float* gsrc = src + lane * 4;
    asm volatile("global_load_async_to_lds_b128 %0, %1, off"
                 :: "v"(lds_off), "v"(gsrc) : "memory");
    asm volatile("s_wait_asynccnt 0" ::: "memory");
    float4 v = *(const float4*)(slot + lane * 4);
    float* dst = agg + (size_t)p * Dd + lane * 4;
    atomicAdd(dst + 0, v.x); atomicAdd(dst + 1, v.y);
    atomicAdd(dst + 2, v.z); atomicAdd(dst + 3, v.w);
}

// --- filter_adj outputs ----------------------------------------------------
__global__ void k_edge_out(const int* __restrict__ row, const int* __restrict__ col,
                           const int* __restrict__ pos,
                           float* __restrict__ o_nrow, float* __restrict__ o_ncol,
                           float* __restrict__ o_mask)
{
    int e = blockIdx.x * blockDim.x + threadIdx.x;
    if (e >= Ee) return;
    int nr = pos[row[e]];
    int nc = pos[col[e]];
    bool m = (nr >= 0) && (nc >= 0);
    o_nrow[e] = m ? (float)nr : -1.0f;
    o_ncol[e] = m ? (float)nc : -1.0f;
    o_mask[e] = m ? 1.0f : 0.0f;
}

// ---------------------------------------------------------------------------
extern "C" void kernel_launch(void* const* d_in, const int* in_sizes, int n_in,
                              void* d_out, int out_size, void* d_ws, size_t ws_size,
                              hipStream_t stream)
{
    const float* x     = (const float*)d_in[0];
    const int*   ei    = (const int*)d_in[1];   // [2,E]: row then col
    const int*   batch = (const int*)d_in[2];
    const int* row = ei;
    const int* col = ei + Ee;

    // workspace carve-out (~3 MB)
    char* p = (char*)d_ws;
    float*    wsum    = (float*)p;     p += Nn * sizeof(float);
    float*    deg     = (float*)p;     p += Nn * sizeof(float);
    unsigned* key     = (unsigned*)p;  p += Nn * sizeof(unsigned);
    unsigned* H       = (unsigned*)p;  p += NBIN * sizeof(unsigned);
    unsigned* Epre    = (unsigned*)p;  p += NBIN * sizeof(unsigned);
    unsigned* cursor  = (unsigned*)p;  p += NBIN * sizeof(unsigned);
    int*      binList = (int*)p;       p += Nn * sizeof(int);
    int*      pos     = (int*)p;       p += Nn * sizeof(int);
    int*      perm_i  = (int*)p;       p += Kk * sizeof(int);

    // output layout (flat float32, tuple return order)
    float* out     = (float*)d_out;
    float* o_agg   = out;                        // K*D
    float* o_nrow  = o_agg  + (size_t)Kk * Dd;   // E
    float* o_ncol  = o_nrow + Ee;                // E
    float* o_mask  = o_ncol + Ee;                // E
    float* o_batch = o_mask + Ee;                // K
    float* o_perm  = o_batch + Kk;               // K

    k_zero     <<<(Nn + 255) / 256, 256, 0, stream>>>(wsum, deg, H);
    k_edge_dot <<<(Ee / 16) / 8, 256, 0, stream>>>(x, row, col, wsum, deg);
    k_score    <<<(Nn + 255) / 256, 256, 0, stream>>>(wsum, deg, key, H);
    k_scan     <<<1, 1024, 0, stream>>>(H, Epre, cursor);
    k_scatter  <<<(Nn + 255) / 256, 256, 0, stream>>>(key, cursor, binList);
    k_rank     <<<(Nn + 255) / 256, 256, 0, stream>>>(key, H, Epre, binList, batch,
                                                      pos, perm_i, o_perm, o_batch);
    k_agg_init <<<(Kk * Dd + 255) / 256, 256, 0, stream>>>(x, perm_i, o_agg);
    k_agg_edges<<<Ee / 8, 256, 0, stream>>>(x, row, col, pos, o_agg);
    k_edge_out <<<(Ee + 255) / 256, 256, 0, stream>>>(row, col, pos,
                                                      o_nrow, o_ncol, o_mask);
}
